// GRU_40286793236888
// MI455X (gfx1250) — matrix operand order repaired and checked by reading, not compiled, not run on hardware
//
#include <hip/hip_runtime.h>
#include <hip/hip_bf16.h>
#include <cstdint>

// ---------------------------------------------------------------------------
// AGCRN GRU cell for MI455X (gfx1250, wave32, WMMA bf16 16x16x32, f32 accum)
// B=64, N=1024, Din=Dout=64, C=128, E_DIM=16, CHEB_K=3
// ---------------------------------------------------------------------------

typedef __attribute__((ext_vector_type(16))) __bf16 bfrag16;  // A/B fragment (8 VGPRs)
typedef __attribute__((ext_vector_type(8)))  __bf16 v8bf;     // 16-byte bf16 vector
typedef __attribute__((ext_vector_type(8)))  float  v8f;      // C/D fragment (8 VGPRs)

#define NN   1024      // nodes
#define BB   64        // batch
#define CC   128       // Din+Dout
#define NCOL (BB*CC)   // 8192 columns of the propagation GEMM
#define EDIM 16

// Build a 16-element bf16 fragment from two 8-half (16B) chunks.
__device__ __forceinline__ bfrag16 ld16(const __bf16* p0, const __bf16* p1) {
  v8bf lo = *(const v8bf*)p0;
  v8bf hi = *(const v8bf*)p1;
  bfrag16 r;
#pragma unroll
  for (int i = 0; i < 8; ++i) { r[i] = lo[i]; r[i + 8] = hi[i]; }
  return r;
}

// CDNA5 async copy: 16B global -> LDS, tracked by ASYNCcnt (no VGPR roundtrip).
__device__ __forceinline__ void async_copy16(uint32_t lds_off, const __bf16* g) {
  asm volatile("global_load_async_to_lds_b128 %0, %1, off"
               :: "v"(lds_off), "v"((uint64_t)(uintptr_t)g)
               : "memory");
}
__device__ __forceinline__ void wait_async0() {
  asm volatile("s_wait_asynccnt 0x0" ::: "memory");
}

// ---------------------------------------------------------------------------
// Kernel 1: A = softmax(relu(E E^T)) per row, stored bf16 (row-major NxN).
// ---------------------------------------------------------------------------
__global__ __launch_bounds__(256) void adj_kernel(const float* __restrict__ E,
                                                  __bf16* __restrict__ A) {
  const int n = blockIdx.x;
  const int tid = threadIdx.x;
  __shared__ float En[EDIM];
  __shared__ float red[256];
  if (tid < EDIM) En[tid] = E[n * EDIM + tid];
  __syncthreads();

  float v[4];
#pragma unroll
  for (int i = 0; i < 4; ++i) {
    const int m = tid + i * 256;
    float d = 0.f;
#pragma unroll
    for (int e = 0; e < EDIM; ++e) d += En[e] * E[m * EDIM + e];
    v[i] = fmaxf(d, 0.0f);  // relu
  }
  float lmax = fmaxf(fmaxf(v[0], v[1]), fmaxf(v[2], v[3]));
  red[tid] = lmax; __syncthreads();
  for (int s = 128; s > 0; s >>= 1) {
    if (tid < s) red[tid] = fmaxf(red[tid], red[tid + s]);
    __syncthreads();
  }
  const float mx = red[0];
  __syncthreads();

  float ex[4], ls = 0.f;
#pragma unroll
  for (int i = 0; i < 4; ++i) { ex[i] = __expf(v[i] - mx); ls += ex[i]; }
  red[tid] = ls; __syncthreads();
  for (int s = 128; s > 0; s >>= 1) {
    if (tid < s) red[tid] += red[tid + s];
    __syncthreads();
  }
  const float inv = 1.0f / red[0];
#pragma unroll
  for (int i = 0; i < 4; ++i) {
    const int m = tid + i * 256;
    A[n * NN + m] = (__bf16)(ex[i] * inv);
  }
}

// ---------------------------------------------------------------------------
// Kernel 2: X0[n,b,c] = bf16(concat(x, state))   layout (N, B, C)
// ---------------------------------------------------------------------------
__global__ __launch_bounds__(256) void concat_in(const float* __restrict__ x,
                                                 const float* __restrict__ st,
                                                 __bf16* __restrict__ X0) {
  const int total = NN * BB * CC;
  for (int idx = blockIdx.x * blockDim.x + threadIdx.x; idx < total;
       idx += gridDim.x * blockDim.x) {
    const int n = idx >> 13;            // /(B*C)
    const int rem = idx & (NCOL - 1);
    const int b = rem >> 7;             // /C
    const int c = rem & 127;
    float v = (c < 64) ? x[(b * NN + n) * 64 + c]
                       : st[(b * NN + n) * 64 + (c - 64)];
    X0[idx] = (__bf16)v;
  }
}

// ---------------------------------------------------------------------------
// Kernel 6: C0[n,b,c] = bf16(concat(x, z*state)), z = zr[n,b,0:64] (sigmoided)
// ---------------------------------------------------------------------------
__global__ __launch_bounds__(256) void concat_cand(const float* __restrict__ x,
                                                   const float* __restrict__ st,
                                                   const float* __restrict__ zr,
                                                   __bf16* __restrict__ C0) {
  const int total = NN * BB * CC;
  for (int idx = blockIdx.x * blockDim.x + threadIdx.x; idx < total;
       idx += gridDim.x * blockDim.x) {
    const int n = idx >> 13;
    const int rem = idx & (NCOL - 1);
    const int b = rem >> 7;
    const int c = rem & 127;
    float v;
    if (c < 64) {
      v = x[(b * NN + n) * 64 + c];
    } else {
      const float z = zr[idx - 64];  // zr shares (n,b,128) layout
      v = z * st[(b * NN + n) * 64 + (c - 64)];
    }
    C0[idx] = (__bf16)v;
  }
}

// ---------------------------------------------------------------------------
// Kernels 3/4/7/8: OUT = alpha*(A @ X) + beta*SUB  (bf16 in, f32 acc, bf16 out)
// A: 1024x1024, X/SUB/OUT: 1024x8192.  Block tile 128x128, BK=32, 8 waves.
// Double-buffered LDS; next tile staged with global_load_async_to_lds_b128
// (ASYNCcnt) so the copy overlaps the 8 v_wmma of the current tile.
// ---------------------------------------------------------------------------
__global__ __launch_bounds__(256) void prop_gemm(const __bf16* __restrict__ Ag,
                                                 const __bf16* __restrict__ Xg,
                                                 const __bf16* __restrict__ Sg,
                                                 __bf16* __restrict__ Og,
                                                 float alpha, float beta) {
  __shared__ __bf16 As[2][128][40];   // 32 K + 8 pad  (80B row stride)
  __shared__ __bf16 Xs[2][32][136];   // 128 N + 8 pad (272B row stride)

  const int tid = threadIdx.x;
  const int lane = tid & 31;
  const int wave = tid >> 5;
  const int wrow0 = (wave >> 2) * 64;   // wave row origin inside 128-tile
  const int wcol0 = (wave & 3) * 32;    // wave col origin inside 128-tile
  const int mbase = blockIdx.y * 128;
  const int nbase = blockIdx.x * 128;
  const int koff = (lane & 16) ? 8 : 0; // A-fragment K split per ISA layout
  const int mrow = lane & 15;

  // Each thread owns two 16B chunks of the A tile and two of the X tile.
  const int i2 = tid + 256;
  const __bf16* gA0 = Ag + (mbase + (tid >> 2)) * NN + (tid & 3) * 8;
  const __bf16* gA1 = Ag + (mbase + (i2  >> 2)) * NN + (i2  & 3) * 8;
  const __bf16* gX0 = Xg + (tid >> 4) * NCOL + nbase + (tid & 15) * 8;
  const __bf16* gX1 = Xg + (i2  >> 4) * NCOL + nbase + (i2  & 15) * 8;
  uint32_t lA0[2], lA1[2], lX0[2], lX1[2];
#pragma unroll
  for (int b = 0; b < 2; ++b) {
    lA0[b] = (uint32_t)(uintptr_t)&As[b][tid >> 2][(tid & 3) * 8];
    lA1[b] = (uint32_t)(uintptr_t)&As[b][i2  >> 2][(i2  & 3) * 8];
    lX0[b] = (uint32_t)(uintptr_t)&Xs[b][tid >> 4][(tid & 15) * 8];
    lX1[b] = (uint32_t)(uintptr_t)&Xs[b][i2  >> 4][(i2  & 15) * 8];
  }

  // Preload tile 0 into buffer 0.
  async_copy16(lA0[0], gA0);
  async_copy16(lA1[0], gA1);
  async_copy16(lX0[0], gX0);
  async_copy16(lX1[0], gX1);
  wait_async0();
  __syncthreads();

  v8f acc[4][2] = {};

  for (int kt = 0; kt < 32; ++kt) {
    const int cur = kt & 1;
    if (kt < 31) {  // stage next K-tile into the other buffer (overlaps WMMA)
      const int nxt = cur ^ 1;
      const int ka = (kt + 1) * 32;          // A advances 32 halfs per step
      const int kx = (kt + 1) * 32 * NCOL;   // X advances 32 rows per step
      async_copy16(lA0[nxt], gA0 + ka);
      async_copy16(lA1[nxt], gA1 + ka);
      async_copy16(lX0[nxt], gX0 + kx);
      async_copy16(lX1[nxt], gX1 + kx);
    }

    bfrag16 af[4], bf[2];
#pragma unroll
    for (int mt = 0; mt < 4; ++mt) {
      const __bf16* p = &As[cur][wrow0 + mt * 16 + mrow][koff];
      af[mt] = ld16(p, p + 16);
    }
#pragma unroll
    for (int nt = 0; nt < 2; ++nt) {
      const __bf16* p = &Xs[cur][0][wcol0 + nt * 16 + mrow];  // lane%16 = N col
#pragma unroll
      for (int s = 0; s < 16; ++s) bf[nt][s] = p[(koff + s + (s >= 8 ? 8 : 0)) * 136];
    }
#pragma unroll
    for (int mt = 0; mt < 4; ++mt)
#pragma unroll
      for (int nt = 0; nt < 2; ++nt)
        acc[mt][nt] = __builtin_amdgcn_wmma_f32_16x16x32_bf16(
            false, af[mt], false, bf[nt], (short)0, acc[mt][nt], false, false);

    if (kt < 31) wait_async0();  // next buffer fully in LDS
    __syncthreads();
  }

  const int rlo = (lane >> 4) * 8;
#pragma unroll
  for (int mt = 0; mt < 4; ++mt) {
#pragma unroll
    for (int nt = 0; nt < 2; ++nt) {
      const int j = nbase + wcol0 + nt * 16 + (lane & 15);
#pragma unroll
      for (int v = 0; v < 8; ++v) {
        const int m = mbase + wrow0 + mt * 16 + v + rlo;
        float val = alpha * acc[mt][nt][v];
        if (beta != 0.0f) val += beta * (float)Sg[m * NCOL + j];
        Og[m * NCOL + j] = (__bf16)val;
      }
    }
  }
}

// ---------------------------------------------------------------------------
// Kernels 5/9: fused per-node projection.
//   per node n: W[n] = sum_e E[n,e]*W_pool[e]  (built in LDS, bf16)
//   out[b,o]   = [X0|xg1|xg2][n] (64x384) @ W[n] (384xO)  + E[n]·b_pool
// Gate (O=128): sigmoid -> zr (f32, layout (N,B,128))
// Update (O=64): tanh -> hc; d_out = r*state + (1-r)*hc
// 3 nodes per block; dynamic LDS holds 3 full per-node weight matrices.
// ---------------------------------------------------------------------------
template <int O>
__global__ __launch_bounds__(256) void proj_kernel(
    const __bf16* __restrict__ xg0, const __bf16* __restrict__ xg1,
    const __bf16* __restrict__ xg2, const float* __restrict__ Wpool,
    const float* __restrict__ bpool, const float* __restrict__ Eg,
    float* __restrict__ zr, const float* __restrict__ state,
    float* __restrict__ out) {
  constexpr int G = 3;
  constexpr int KC = 3 * CC;        // 384
  constexpr int OPAD = O + 8;       // keep 16B alignment, spread banks
  constexpr int NT = O / 64;        // WMMA N-tiles per wave (2 gate, 1 upd)
  constexpr bool GATE = (O == 128);
  constexpr int O4 = O / 4;

  extern __shared__ char smem[];
  __bf16* Wlds = (__bf16*)smem;     // [G][KC][OPAD]
  __shared__ float Es[G][EDIM];

  const int tid = threadIdx.x;
  const int n0 = blockIdx.x * G;
  if (tid < G * EDIM) {
    const int g = tid >> 4, e = tid & 15;
    const int n = n0 + g;
    Es[g][e] = (n < NN) ? Eg[n * EDIM + e] : 0.0f;
  }
  __syncthreads();

  // ---- Phase 1: build per-node weights in LDS (bf16), float4 vectorized ----
  for (int idx = tid; idx < KC * O4; idx += 256) {
    const int kc = idx / O4;
    const int o4 = (idx - kc * O4) * 4;
    const float4* wp = (const float4*)(Wpool + kc * O + o4);
    float4 w[EDIM];
#pragma unroll
    for (int e = 0; e < EDIM; ++e) w[e] = wp[e * (KC * O / 4)];
#pragma unroll
    for (int g = 0; g < G; ++g) {
      float sx = 0.f, sy = 0.f, sz = 0.f, sw = 0.f;
#pragma unroll
      for (int e = 0; e < EDIM; ++e) {
        const float c = Es[g][e];
        sx += c * w[e].x; sy += c * w[e].y; sz += c * w[e].z; sw += c * w[e].w;
      }
      __bf16* dst = Wlds + (g * KC + kc) * OPAD + o4;
      dst[0] = (__bf16)sx; dst[1] = (__bf16)sy;
      dst[2] = (__bf16)sz; dst[3] = (__bf16)sw;
    }
  }
  __syncthreads();

  // ---- Phase 2: WMMA (64 x 384) @ (384 x O) per node ----------------------
  const int lane = tid & 31;
  const int wave = tid >> 5;
  const int wrow0 = (wave >> 2) * 32;       // batch rows per wave: 32
  const int wcol0 = (wave & 3) * (16 * NT); // output cols per wave
  const int koff = (lane & 16) ? 8 : 0;
  const int mrow = lane & 15;
  const int rlo = (lane >> 4) * 8;

  for (int g = 0; g < G; ++g) {
    const int n = n0 + g;
    if (n >= NN) break;
    const __bf16* s0 = xg0 + n * NCOL;
    const __bf16* s1 = xg1 + n * NCOL;
    const __bf16* s2 = xg2 + n * NCOL;

    v8f acc[2][NT] = {};
    for (int ks = 0; ks < 12; ++ks) {          // K = 384 = 12 * 32
      const int slab = ks >> 2;
      const int coff = (ks & 3) * 32;
      const __bf16* sp = (slab == 0) ? s0 : (slab == 1) ? s1 : s2;

      bfrag16 af[2], bf[NT];
#pragma unroll
      for (int mt = 0; mt < 2; ++mt) {
        const __bf16* p = sp + (wrow0 + mt * 16 + mrow) * CC + coff + koff;
        af[mt] = ld16(p, p + 16);
      }
#pragma unroll
      for (int nt = 0; nt < NT; ++nt) {
        const __bf16* p = Wlds + (g * KC + ks * 32) * OPAD + wcol0 + nt * 16 + mrow;
#pragma unroll
        for (int s = 0; s < 16; ++s) bf[nt][s] = p[(koff + s + (s >= 8 ? 8 : 0)) * OPAD];
      }
#pragma unroll
      for (int mt = 0; mt < 2; ++mt)
#pragma unroll
        for (int nt = 0; nt < NT; ++nt)
          acc[mt][nt] = __builtin_amdgcn_wmma_f32_16x16x32_bf16(
              false, af[mt], false, bf[nt], (short)0, acc[mt][nt], false, false);
    }

    // ---- epilogue ----------------------------------------------------------
#pragma unroll
    for (int nt = 0; nt < NT; ++nt) {
      const int o = wcol0 + nt * 16 + (lane & 15);
      float bias = 0.f;
#pragma unroll
      for (int e = 0; e < EDIM; ++e) bias += Es[g][e] * bpool[e * O + o];
#pragma unroll
      for (int mt = 0; mt < 2; ++mt) {
#pragma unroll
        for (int v = 0; v < 8; ++v) {
          const int b = wrow0 + mt * 16 + v + rlo;
          const float val = acc[mt][nt][v] + bias;
          if constexpr (GATE) {
            zr[n * NCOL + b * CC + o] = 1.0f / (1.0f + __expf(-val));
          } else {
            const float hc = tanhf(val);
            const float r = zr[n * NCOL + b * CC + 64 + o];
            const float st = state[(b * NN + n) * 64 + o];
            out[(b * NN + n) * 64 + o] = r * st + (1.0f - r) * hc;
          }
        }
      }
    }
  }
}

// ---------------------------------------------------------------------------
// Host-side orchestration
// ---------------------------------------------------------------------------
extern "C" void kernel_launch(void* const* d_in, const int* in_sizes, int n_in,
                              void* d_out, int out_size, void* d_ws, size_t ws_size,
                              hipStream_t stream) {
  const float* x  = (const float*)d_in[0];  // (64,1024,64)
  const float* st = (const float*)d_in[1];  // (64,1024,64)
  const float* E  = (const float*)d_in[2];  // (1024,16)
  const float* gW = (const float*)d_in[3];  // (16,3,128,128)
  const float* gb = (const float*)d_in[4];  // (16,128)
  const float* uW = (const float*)d_in[5];  // (16,3,128,64)
  const float* ub = (const float*)d_in[6];  // (16,64)
  float* out = (float*)d_out;               // (64,1024,64)

  // Workspace layout (~86 MB)
  char* w = (char*)d_ws;
  __bf16* Abf = (__bf16*)w; w += (size_t)NN * NN * 2;          // 2 MB
  __bf16* X0  = (__bf16*)w; w += (size_t)NN * NCOL * 2;        // 16.8 MB
  __bf16* Xg1 = (__bf16*)w; w += (size_t)NN * NCOL * 2;        // 16.8 MB
  __bf16* Xg2 = (__bf16*)w; w += (size_t)NN * NCOL * 2;        // 16.8 MB
  float*  zr  = (float*)w;                                      // 33.5 MB

  const dim3 gemmGrid(NCOL / 128, NN / 128);                   // 64 x 8
  const int projBlocks = (NN + 2) / 3;                          // 342
  const int gateShmem = 3 * 384 * (128 + 8) * 2;               // 313,344 B
  const int updShmem  = 3 * 384 * (64 + 8) * 2;                // 165,888 B

  // 1. adjacency
  adj_kernel<<<NN, 256, 0, stream>>>(E, Abf);
  // 2. X0 = bf16(concat(x, state))            (N,B,C)
  concat_in<<<4096, 256, 0, stream>>>(x, st, X0);
  // 3. xg1 = A @ X0
  prop_gemm<<<gemmGrid, 256, 0, stream>>>(Abf, X0, X0, Xg1, 1.0f, 0.0f);
  // 4. xg2 = 2*A @ xg1 - X0                    (= T2 @ X0)
  prop_gemm<<<gemmGrid, 256, 0, stream>>>(Abf, Xg1, X0, Xg2, 2.0f, -1.0f);
  // 5. gate projection -> zr = sigmoid(...)
  proj_kernel<128><<<projBlocks, 256, gateShmem, stream>>>(
      X0, Xg1, Xg2, gW, gb, E, zr, nullptr, nullptr);
  // 6. C0 = bf16(concat(x, z*state))   (reuse X0 buffer)
  concat_cand<<<4096, 256, 0, stream>>>(x, st, zr, X0);
  // 7. cg1 = A @ C0
  prop_gemm<<<gemmGrid, 256, 0, stream>>>(Abf, X0, X0, Xg1, 1.0f, 0.0f);
  // 8. cg2 = 2*A @ cg1 - C0
  prop_gemm<<<gemmGrid, 256, 0, stream>>>(Abf, Xg1, X0, Xg2, 2.0f, -1.0f);
  // 9. update projection -> out = r*state + (1-r)*tanh(...)
  proj_kernel<64><<<projBlocks, 256, updShmem, stream>>>(
      X0, Xg1, Xg2, uW, ub, E, zr, st, out);
}
